// VisionTransformer_51591147159825
// MI455X (gfx1250) — matrix operand order repaired and checked by reading, not compiled, not run on hardware
//
#include <hip/hip_runtime.h>
#include <hip/hip_bf16.h>
#include <math.h>

typedef __bf16 bf16;
typedef __attribute__((ext_vector_type(16))) __bf16 v16bf;
typedef __attribute__((ext_vector_type(8)))  float  v8f;

#define D_    768
#define S_    196
#define SP_   224      // padded sequence (multiple of 32)
#define H_    12
#define F_    3072
#define NB_   32
#define NC_   1000
#define L_    12
#define MTOK_ (NB_*S_) // 6272, multiple of 128

enum { EPI_H=0, EPI_TOPAD, EPI_VT, EPI_SCORES, EPI_AV, EPI_GELU, EPI_RES, EPI_HEAD };

// ---------------------------------------------------------------------------
// Core bf16 WMMA GEMM: C = A[M,K] * Bt[N,K]^T, fp32 accumulate, epilogue by EPI.
// Block tile 128(M) x 128(N), K step 32. 8 waves of 32 arranged 4(M) x 2(N);
// each wave owns 32x64 = 2x4 WMMA accumulators. LDS double-buffered: one
// barrier per K step; staging of tile k+1 overlaps WMMA of tile k.
// ---------------------------------------------------------------------------
template<int EPI>
__global__ __launch_bounds__(256) void gemm_wmma_bf16(
    const bf16* __restrict__ A, int lda,
    const bf16* __restrict__ Bt, int ldb,
    int M, int N, int K, int Mout, int Nout, int ldc,
    const float* __restrict__ bias, const float* __restrict__ pos,
    float* __restrict__ outF, bf16* __restrict__ outB,
    float* __restrict__ res)
{
  constexpr int BM = 128, BN = 128, BK = 32;
  __shared__ __align__(16) bf16 As[2][BM * BK];   // 2 x 8 KB
  __shared__ __align__(16) bf16 Bs[2][BN * BK];   // 2 x 8 KB

  const int tid  = threadIdx.x;
  const int wave = tid >> 5, lane = tid & 31;
  const int wm = (wave & 3) * 32;        // wave M offset inside block tile
  const int wn = (wave >> 2) * 64;       // wave N offset inside block tile
  const int bm = blockIdx.y * BM;
  const int bn = blockIdx.x * BN;
  const int batch = blockIdx.z;

  long long aoff = 0, boff = 0;
  if constexpr (EPI == EPI_SCORES) {
    const int b = batch / H_, hh = batch % H_;
    aoff = boff = (long long)b * SP_ * D_ + hh * 64;
  } else if constexpr (EPI == EPI_AV) {
    aoff = (long long)batch * SP_ * SP_;
    boff = (long long)batch * 64 * SP_;
  }
  A  += aoff;
  Bt += boff;

  // staging: 256 threads, each moves 16 bf16 (2 x b128) of A and of B
  const int srow = tid >> 1;            // 0..127
  const int scol = (tid & 1) * 16;      // 0 or 16
  const bool aok = (bm + srow) < M;
  const bool bok = (bn + srow) < N;
  const bf16* arowp = A  + (long long)(bm + srow) * lda + scol;
  const bf16* browp = Bt + (long long)(bn + srow) * ldb + scol;

  auto stage = [&](int k0, int buf) {
    uint4 a0 = {0u,0u,0u,0u}, a1 = {0u,0u,0u,0u};
    uint4 b0 = {0u,0u,0u,0u}, b1 = {0u,0u,0u,0u};
    if (aok) {
      a0 = *(const uint4*)(arowp + k0);
      a1 = *(const uint4*)(arowp + k0 + 8);
    }
    if (bok) {
      b0 = *(const uint4*)(browp + k0);
      b1 = *(const uint4*)(browp + k0 + 8);
    }
    *(uint4*)(As[buf] + srow * BK + scol)     = a0;
    *(uint4*)(As[buf] + srow * BK + scol + 8) = a1;
    *(uint4*)(Bs[buf] + srow * BK + scol)     = b0;
    *(uint4*)(Bs[buf] + srow * BK + scol + 8) = b1;
  };

  const int hf  = lane >> 4;   // 0: lanes 0-15, 1: lanes 16-31
  const int r16 = lane & 15;

  union Frag { v16bf v; uint4 q[2]; };
  v8f acc[2][4] = {};

  stage(0, 0);
  __syncthreads();

  int buf = 0;
  for (int k0 = 0; k0 < K; k0 += BK) {
    // stage next tile into the other buffer (overlaps with WMMA below)
    if (k0 + BK < K) stage(k0 + BK, buf ^ 1);
    // prefetch two tiles ahead into caches (global_prefetch_b8)
    if (k0 + 2 * BK < K) {
      if (aok) __builtin_prefetch(arowp + k0 + 2 * BK, 0, 1);
      if (bok) __builtin_prefetch(browp + k0 + 2 * BK, 0, 1);
    }

    // A fragment (16x32 bf16): lanes 0-15 row r K0..7 & K16..23; lanes 16-31 K8..15 & K24..31
    Frag af[2], bfr[4];
#pragma unroll
    for (int i = 0; i < 2; ++i) {
      const bf16* pa = As[buf] + (wm + i * 16 + r16) * BK + hf * 8;
      af[i].q[0] = *(const uint4*)(pa);
      af[i].q[1] = *(const uint4*)(pa + 16);
    }
    // B fragment (32x16 bf16): lanes 0-15 col c K0..7 & K16..23; lanes 16-31 K8..15 & K24..31
#pragma unroll
    for (int j = 0; j < 4; ++j) {
      const bf16* pb = Bs[buf] + (wn + j * 16 + r16) * BK + hf * 8;
      bfr[j].q[0] = *(const uint4*)(pb);
      bfr[j].q[1] = *(const uint4*)(pb + 16);
    }
#pragma unroll
    for (int i = 0; i < 2; ++i)
#pragma unroll
      for (int j = 0; j < 4; ++j)
        acc[i][j] = __builtin_amdgcn_wmma_f32_16x16x32_bf16(
            false, af[i].v, false, bfr[j].v, (short)0, acc[i][j], false, false);

    __syncthreads();   // next tile fully staged; this buffer's reads done
    buf ^= 1;
  }

  // Epilogue. C/D layout: VGPR a: lanes 0-15 -> (M=a, N=lane); lanes 16-31 -> (M=8+a).
  const int rbase = bm + wm + hf * 8;
  const int cbase = bn + wn + r16;
#pragma unroll
  for (int i = 0; i < 2; ++i) {
#pragma unroll
    for (int j = 0; j < 4; ++j) {
      const int col = cbase + j * 16;
      if (col >= Nout) continue;
      const float* av = (const float*)&acc[i][j];
#pragma unroll
      for (int a = 0; a < 8; ++a) {
        const int row = rbase + i * 16 + a;
        if (row >= Mout) continue;
        float v = av[a];
        if constexpr (EPI == EPI_H) {
          const int s = row % S_;
          outF[(long long)row * D_ + col] = v + bias[col] + pos[(long long)s * D_ + col];
        } else if constexpr (EPI == EPI_TOPAD) {
          const int b = row / S_, s = row % S_;
          outB[((long long)b * SP_ + s) * D_ + col] = (bf16)v;
        } else if constexpr (EPI == EPI_VT) {
          const int b = row / S_, s = row % S_;
          const int hh = col >> 6, dh = col & 63;
          outB[(((long long)b * H_ + hh) * 64 + dh) * SP_ + s] = (bf16)v;
        } else if constexpr (EPI == EPI_SCORES) {
          outF[(long long)batch * SP_ * SP_ + (long long)row * SP_ + col] = v;
        } else if constexpr (EPI == EPI_AV) {
          const int b = batch / H_, hh = batch % H_;
          res[((long long)(b * S_ + row)) * D_ + hh * 64 + col] += v;
        } else if constexpr (EPI == EPI_GELU) {
          const float u = v + bias[col];
          const float g = 0.5f * u * (1.0f + erff(u * 0.70710678118654752f));
          outB[(long long)row * ldc + col] = (bf16)g;
        } else if constexpr (EPI == EPI_RES) {
          res[(long long)row * D_ + col] += v + bias[col];
        } else { // EPI_HEAD
          outF[(long long)row * ldc + col] = v + bias[col];
        }
      }
    }
  }
}

// ---------------------------------------------------------------------------
// Helpers
// ---------------------------------------------------------------------------
__global__ __launch_bounds__(256) void f32_to_bf16_k(const float* __restrict__ in,
                                                     bf16* __restrict__ out, long long n) {
  const long long i = (long long)blockIdx.x * 256 + threadIdx.x;
  if (i < n) out[i] = (bf16)in[i];
}

// out[n*K + k] = in[k*N + n]  (zero-pad rows n >= N up to Nalloc)
__global__ __launch_bounds__(256) void transpose_to_bf16_k(const float* __restrict__ in,
                                                           bf16* __restrict__ out,
                                                           int K, int N, int Nalloc) {
  const long long i = (long long)blockIdx.x * 256 + threadIdx.x;
  const long long total = (long long)Nalloc * K;
  if (i >= total) return;
  const int k = (int)(i % K);
  const int n = (int)(i / K);
  out[i] = (n < N) ? (bf16)in[(long long)k * N + n] : (bf16)0.f;
}

__global__ __launch_bounds__(256) void zero_bf16_k(bf16* __restrict__ p, long long n) {
  const long long i = (long long)blockIdx.x * 256 + threadIdx.x;
  if (i < n) p[i] = (bf16)0.f;
}

__global__ __launch_bounds__(256) void im2col_k(const float* __restrict__ x,
                                                bf16* __restrict__ a) {
  const long long idx = (long long)blockIdx.x * 256 + threadIdx.x;
  if (idx >= (long long)MTOK_ * D_) return;
  const int kk = (int)(idx % D_);
  const long long rowi = idx / D_;
  const int b = (int)(rowi / S_), s = (int)(rowi % S_);
  const int ph = s / 14, pw = s % 14;
  const int c = kk >> 8, rem = kk & 255;
  const int ii = rem >> 4, jj = rem & 15;
  a[idx] = (bf16)x[(((long long)b * 3 + c) * 224 + ph * 16 + ii) * 224 + pw * 16 + jj];
}

__global__ __launch_bounds__(256) void layernorm_bf16_k(const float* __restrict__ x,
                                                        const float* __restrict__ gamma,
                                                        const float* __restrict__ beta,
                                                        bf16* __restrict__ out) {
  const int row = blockIdx.x;
  const float* xr = x + (long long)row * D_;
  const int t = threadIdx.x;
  const float a0 = xr[t], a1 = xr[t + 256], a2 = xr[t + 512];
  __shared__ float red[8];
  float s = a0 + a1 + a2;
  for (int m = 16; m; m >>= 1) s += __shfl_xor(s, m, 32);
  if ((t & 31) == 0) red[t >> 5] = s;
  __syncthreads();
  const float mean = (red[0]+red[1]+red[2]+red[3]+red[4]+red[5]+red[6]+red[7]) * (1.0f / D_);
  __syncthreads();
  const float d0 = a0 - mean, d1 = a1 - mean, d2 = a2 - mean;
  float vv = d0*d0 + d1*d1 + d2*d2;
  for (int m = 16; m; m >>= 1) vv += __shfl_xor(vv, m, 32);
  if ((t & 31) == 0) red[t >> 5] = vv;
  __syncthreads();
  const float var = (red[0]+red[1]+red[2]+red[3]+red[4]+red[5]+red[6]+red[7]) * (1.0f / D_);
  const float rs = rsqrtf(var + 1e-5f);
  bf16* o = out + (long long)row * D_;
  o[t]       = (bf16)(d0 * rs * gamma[t]       + beta[t]);
  o[t + 256] = (bf16)(d1 * rs * gamma[t + 256] + beta[t + 256]);
  o[t + 512] = (bf16)(d2 * rs * gamma[t + 512] + beta[t + 512]);
}

__global__ __launch_bounds__(256) void softmax_k(const float* __restrict__ scores,
                                                 bf16* __restrict__ att) {
  const long long row = blockIdx.x;          // B*H*SP rows
  const int s = (int)(row % SP_);
  const float* src = scores + row * SP_;
  bf16* dst = att + row * SP_;
  const int t = threadIdx.x;
  if (s >= S_) {                              // padded query rows -> zeros (uniform branch)
    if (t < SP_) dst[t] = (bf16)0.f;
    return;
  }
  const float scale = 0.0714285714285714f;    // 1/sqrt(196) = 1/14
  const float v = (t < S_) ? src[t] * scale : -3.4e38f;
  __shared__ float red[8];
  float m = v;
  for (int o = 16; o; o >>= 1) m = fmaxf(m, __shfl_xor(m, o, 32));
  if ((t & 31) == 0) red[t >> 5] = m;
  __syncthreads();
  m = fmaxf(fmaxf(fmaxf(red[0], red[1]), fmaxf(red[2], red[3])),
            fmaxf(fmaxf(red[4], red[5]), fmaxf(red[6], red[7])));
  __syncthreads();
  const float e = (t < S_) ? __expf(v - m) : 0.f;
  float sum = e;
  for (int o = 16; o; o >>= 1) sum += __shfl_xor(sum, o, 32);
  if ((t & 31) == 0) red[t >> 5] = sum;
  __syncthreads();
  sum = red[0]+red[1]+red[2]+red[3]+red[4]+red[5]+red[6]+red[7];
  const float inv = 1.0f / sum;
  if (t < SP_) dst[t] = (bf16)((t < S_) ? e * inv : 0.f);
}

__global__ __launch_bounds__(256) void meanpool_k(const float* __restrict__ h,
                                                  bf16* __restrict__ pooled) {
  const int b = blockIdx.x;
  const int d = blockIdx.y * 256 + threadIdx.x;   // 0..767
  float sum = 0.f;
  for (int s = 0; s < S_; ++s) sum += h[((long long)b * S_ + s) * D_ + d];
  pooled[(long long)b * D_ + d] = (bf16)(sum * (1.0f / S_));
}

// ---------------------------------------------------------------------------
// Orchestration
// ---------------------------------------------------------------------------
static inline unsigned cdivu(long long a, long long b) { return (unsigned)((a + b - 1) / b); }

extern "C" void kernel_launch(void* const* d_in, const int* in_sizes, int n_in,
                              void* d_out, int out_size, void* d_ws, size_t ws_size,
                              hipStream_t stream) {
  (void)in_sizes; (void)n_in; (void)out_size; (void)ws_size;
  const float* x      = (const float*)d_in[0];
  const float* conv_w = (const float*)d_in[1];
  const float* conv_b = (const float*)d_in[2];
  const float* pos    = (const float*)d_in[3];
  const float* ln1_s  = (const float*)d_in[4];
  const float* ln1_b  = (const float*)d_in[5];
  const float* wq     = (const float*)d_in[6];
  const float* wk     = (const float*)d_in[7];
  const float* wv     = (const float*)d_in[8];
  const float* ln2_s  = (const float*)d_in[9];
  const float* ln2_b  = (const float*)d_in[10];
  const float* w1     = (const float*)d_in[11];
  const float* b1     = (const float*)d_in[12];
  const float* w2     = (const float*)d_in[13];
  const float* b2     = (const float*)d_in[14];
  const float* fc_w   = (const float*)d_in[15];
  const float* fc_b   = (const float*)d_in[16];
  float* out = (float*)d_out;

  char* p = (char*)d_ws;
  auto carve = [&](size_t bytes) -> void* {
    void* r = (void*)p;
    p += (bytes + 255) & ~(size_t)255;
    return r;
  };

  bf16*  cw_bf  = (bf16*) carve((size_t)D_ * D_ * 2);
  bf16*  wqT    = (bf16*) carve((size_t)D_ * D_ * 2);
  bf16*  wkT    = (bf16*) carve((size_t)D_ * D_ * 2);
  bf16*  wvT    = (bf16*) carve((size_t)D_ * D_ * 2);
  bf16*  w1T    = (bf16*) carve((size_t)D_ * F_ * 2);
  bf16*  w2T    = (bf16*) carve((size_t)F_ * D_ * 2);
  bf16*  fcT    = (bf16*) carve((size_t)1024 * D_ * 2);
  bf16*  Ape    = (bf16*) carve((size_t)MTOK_ * D_ * 2);
  float* hbuf   = (float*)carve((size_t)MTOK_ * D_ * 4);
  bf16*  xn     = (bf16*) carve((size_t)MTOK_ * D_ * 2);
  bf16*  qpad   = (bf16*) carve((size_t)NB_ * SP_ * D_ * 2);
  bf16*  kpad   = (bf16*) carve((size_t)NB_ * SP_ * D_ * 2);
  bf16*  vT     = (bf16*) carve((size_t)NB_ * H_ * 64 * SP_ * 2);
  float* scores = (float*)carve((size_t)NB_ * H_ * SP_ * SP_ * 4);
  bf16*  att    = (bf16*) carve((size_t)NB_ * H_ * SP_ * SP_ * 2);
  bf16*  ybuf   = (bf16*) carve((size_t)MTOK_ * F_ * 2);
  bf16*  pooled = (bf16*) carve((size_t)NB_ * D_ * 2);

  const dim3 blk(256);

  // layer-invariant weight prep
  f32_to_bf16_k<<<cdivu((long long)D_ * D_, 256), blk, 0, stream>>>(conv_w, cw_bf, (long long)D_ * D_);
  transpose_to_bf16_k<<<cdivu((long long)1024 * D_, 256), blk, 0, stream>>>(fc_w, fcT, D_, NC_, 1024);

  // im2col patches -> bf16 A
  im2col_k<<<cdivu((long long)MTOK_ * D_, 256), blk, 0, stream>>>(x, Ape);

  // zero padded activation buffers (pad rows/cols stay zero all layers)
  zero_bf16_k<<<cdivu((long long)NB_ * SP_ * D_, 256), blk, 0, stream>>>(qpad, (long long)NB_ * SP_ * D_);
  zero_bf16_k<<<cdivu((long long)NB_ * SP_ * D_, 256), blk, 0, stream>>>(kpad, (long long)NB_ * SP_ * D_);
  zero_bf16_k<<<cdivu((long long)NB_ * H_ * 64 * SP_, 256), blk, 0, stream>>>(vT, (long long)NB_ * H_ * 64 * SP_);

  // patch embedding: h = Ape @ conv_w^T + conv_b + pos_embed
  gemm_wmma_bf16<EPI_H><<<dim3(D_ / 128, MTOK_ / 128, 1), blk, 0, stream>>>(
      Ape, D_, cw_bf, D_, MTOK_, D_, D_, MTOK_, D_, D_,
      conv_b, pos, hbuf, nullptr, nullptr);

  for (int l = 0; l < L_; ++l) {
    // per-layer weight conversion (transpose to [N,K] bf16)
    transpose_to_bf16_k<<<cdivu((long long)D_ * D_, 256), blk, 0, stream>>>(wq + (size_t)l * D_ * D_, wqT, D_, D_, D_);
    transpose_to_bf16_k<<<cdivu((long long)D_ * D_, 256), blk, 0, stream>>>(wk + (size_t)l * D_ * D_, wkT, D_, D_, D_);
    transpose_to_bf16_k<<<cdivu((long long)D_ * D_, 256), blk, 0, stream>>>(wv + (size_t)l * D_ * D_, wvT, D_, D_, D_);
    transpose_to_bf16_k<<<cdivu((long long)F_ * D_, 256), blk, 0, stream>>>(w1 + (size_t)l * D_ * F_, w1T, D_, F_, F_);
    transpose_to_bf16_k<<<cdivu((long long)D_ * F_, 256), blk, 0, stream>>>(w2 + (size_t)l * F_ * D_, w2T, F_, D_, D_);

    // LN1
    layernorm_bf16_k<<<MTOK_, blk, 0, stream>>>(hbuf, ln1_s + (size_t)l * D_, ln1_b + (size_t)l * D_, xn);

    // Q, K -> padded [B,SP,D]; V -> transposed [B,H,64,SP]
    gemm_wmma_bf16<EPI_TOPAD><<<dim3(D_ / 128, MTOK_ / 128, 1), blk, 0, stream>>>(
        xn, D_, wqT, D_, MTOK_, D_, D_, MTOK_, D_, D_, nullptr, nullptr, nullptr, qpad, nullptr);
    gemm_wmma_bf16<EPI_TOPAD><<<dim3(D_ / 128, MTOK_ / 128, 1), blk, 0, stream>>>(
        xn, D_, wkT, D_, MTOK_, D_, D_, MTOK_, D_, D_, nullptr, nullptr, nullptr, kpad, nullptr);
    gemm_wmma_bf16<EPI_VT><<<dim3(D_ / 128, MTOK_ / 128, 1), blk, 0, stream>>>(
        xn, D_, wvT, D_, MTOK_, D_, D_, MTOK_, D_, D_, nullptr, nullptr, nullptr, vT, nullptr);

    // scores[b,h] = Q @ K^T  (batched over B*H)
    gemm_wmma_bf16<EPI_SCORES><<<dim3(cdivu(SP_, 128), cdivu(SP_, 128), NB_ * H_), blk, 0, stream>>>(
        qpad, D_, kpad, D_, SP_, SP_, 64, SP_, SP_, SP_, nullptr, nullptr, scores, nullptr, nullptr);

    // softmax rows (scale 1/14), bf16 out, zero padding
    softmax_k<<<NB_ * H_ * SP_, blk, 0, stream>>>(scores, att);

    // h += att @ V  (batched; accumulate into residual)
    gemm_wmma_bf16<EPI_AV><<<dim3(1, cdivu(SP_, 128), NB_ * H_), blk, 0, stream>>>(
        att, SP_, vT, SP_, SP_, 64, SP_, S_, 64, 0, nullptr, nullptr, nullptr, nullptr, hbuf);

    // LN2
    layernorm_bf16_k<<<MTOK_, blk, 0, stream>>>(hbuf, ln2_s + (size_t)l * D_, ln2_b + (size_t)l * D_, xn);

    // MLP1: y = gelu(xn @ W1 + b1)
    gemm_wmma_bf16<EPI_GELU><<<dim3(F_ / 128, MTOK_ / 128, 1), blk, 0, stream>>>(
        xn, D_, w1T, D_, MTOK_, F_, D_, MTOK_, F_, F_, b1 + (size_t)l * F_, nullptr, nullptr, ybuf, nullptr);

    // MLP2: h += y @ W2 + b2
    gemm_wmma_bf16<EPI_RES><<<dim3(D_ / 128, MTOK_ / 128, 1), blk, 0, stream>>>(
        ybuf, F_, w2T, F_, MTOK_, D_, F_, MTOK_, D_, D_, b2 + (size_t)l * D_, nullptr, nullptr, nullptr, hbuf);
  }

  // classifier head
  meanpool_k<<<dim3(NB_, 3), blk, 0, stream>>>(hbuf, pooled);
  gemm_wmma_bf16<EPI_HEAD><<<dim3(1024 / 128, cdivu(NB_, 128), 1), blk, 0, stream>>>(
      pooled, D_, fcT, D_, NB_, 1024, D_, NB_, NC_, NC_, fc_b, nullptr, out, nullptr, nullptr);
}
